// FastAttention_72885595013373
// MI455X (gfx1250) — matrix operand order repaired and checked by reading, not compiled, not run on hardware
//
#include <hip/hip_runtime.h>
#include <hip/hip_bf16.h>
#include <stdint.h>

// ---------------- Problem constants ----------------
#define BATCH   4
#define SEQ     8192
#define DIM     1024
#define HEADS   8
#define DH      64
#define INNER   512           // HEADS*DH
#define QKV3    1536          // 3*INNER
#define MTOK    32768         // BATCH*SEQ

// ---------------- WMMA fragment types ----------------
typedef __attribute__((ext_vector_type(16))) __bf16 v16bf;
typedef __attribute__((ext_vector_type(8)))  float  v8f;
typedef __attribute__((ext_vector_type(4)))  unsigned int v4u;
typedef __attribute__((ext_vector_type(8)))  int v8i;
typedef __attribute__((ext_vector_type(4)))  int v4i;

union FragAB { uint32_t u[8]; v16bf v; };
union FragC  { float f[8]; v8f v; };

__device__ __forceinline__ uint16_t f32_to_bf16(float f) {
    uint32_t u = __float_as_uint(f);
    uint32_t r = (u + 0x7FFFu + ((u >> 16) & 1u)) >> 16;   // round-nearest-even
    return (uint16_t)r;
}

// ---------------- Convert kernels ----------------
__global__ __launch_bounds__(256) void cvt_bf16(const float* __restrict__ in,
                                                uint16_t* __restrict__ out, int n) {
    for (int i = blockIdx.x * blockDim.x + threadIdx.x; i < n; i += gridDim.x * blockDim.x)
        out[i] = f32_to_bf16(in[i]);
}

// in: [K,N] row-major f32  ->  out: [N,K] row-major bf16
__global__ __launch_bounds__(256) void cvt_bf16_T(const float* __restrict__ in,
                                                  uint16_t* __restrict__ out, int K, int N) {
    int total = K * N;
    for (int i = blockIdx.x * blockDim.x + threadIdx.x; i < total; i += gridDim.x * blockDim.x) {
        int k = i / N, n = i - k * N;
        out[(size_t)n * K + k] = f32_to_bf16(in[i]);
    }
}

// ---------------- Tensor Data Mover: 2D bf16 tile -> LDS (padded rows) ----------------
// D# per CDNA5 ISA ch.8: group0 = {flags, lds_addr, global_addr[56:0], type=2},
// group1 = {data_size=2B, pad 16DW interval / 4DW amount, dims/strides, tile dims}.
// Row in LDS: 64B data + 16B pad = 80B stride == sX[r][BK+8] layout.
#define TDM_AVAILABLE __has_builtin(__builtin_amdgcn_tensor_load_to_lds)

#if TDM_AVAILABLE
__device__ __forceinline__ void tdm_load_tile_2d(uint32_t lds_off,
                                                 const uint16_t* gsrc,
                                                 uint32_t tile_w,      // elements (bf16)
                                                 uint32_t tile_h,      // rows
                                                 uint32_t stride_el) { // row stride, elements
    uint64_t ga = (uint64_t)(uintptr_t)gsrc;
    v4u g0;
    g0[0] = 1u;                                            // count=1, user descriptor
    g0[1] = lds_off;                                       // lds_addr (bytes)
    g0[2] = (uint32_t)ga;                                  // global_addr[31:0]
    g0[3] = ((uint32_t)(ga >> 32) & 0x01FFFFFFu)           // global_addr[56:32]
            | 0x80000000u;                                 // type=2 ("image")
    uint32_t dim0 = stride_el;                             // tensor_dim0 >= tile_w
    uint32_t dim1 = tile_h;                                // tensor_dim1 >= tile_h
    v8i g1;
    g1[0] = (int)((1u << 16)      // data_size = 2 bytes
                | (1u << 20)      // pad_enable
                | (3u << 22)      // pad_interval: 16 DWORDs (64B) between pads
                | (3u << 25));    // pad_amount:   4 DWORDs (16B)
    g1[1] = (int)((dim0 & 0xFFFFu) << 16);                 // [47:32]=0, [63:48]=dim0.lo
    g1[2] = (int)((dim0 >> 16) | ((dim1 & 0xFFFFu) << 16));// dim0.hi, dim1.lo
    g1[3] = (int)((dim1 >> 16) | (tile_w << 16));          // dim1.hi, tile_dim0
    g1[4] = (int)tile_h;                                   // tile_dim1 (tile_dim2=0 -> 2D)
    g1[5] = (int)stride_el;                                // tensor_dim0_stride[31:0]
    g1[6] = 0;                                             // stride hi, dim1_stride lo
    g1[7] = 0;
    v4i gz4 = {0, 0, 0, 0};                                // groups 2/3 unused (2D)
    v8i gz8 = {0, 0, 0, 0, 0, 0, 0, 0};                    // extra group (clang-23 form)
    __builtin_amdgcn_tensor_load_to_lds(g0, g1, gz4, gz4, gz8, 0);
}
#endif

// ---------------- Tiled NT GEMM: C[M,N] = A[M,K] * Bt[N,K]^T (+bias) ----------------
#define BM 128
#define BN 128
#define BK 32
#define APAD 8   // 8 bf16 = 16B pad; row stride 80B (16B aligned, bank-spread)

__global__ __launch_bounds__(256) void gemm_bf16_nt(const uint16_t* __restrict__ A,
                                                    const uint16_t* __restrict__ Bt,
                                                    const float* __restrict__ bias,
                                                    float* __restrict__ C,
                                                    int M, int N, int K) {
    __shared__ __align__(16) uint16_t sA[2][BM][BK + APAD];   // double-buffered
    __shared__ __align__(16) uint16_t sB[2][BN][BK + APAD];

    const int tid  = threadIdx.x;
    const int lane = tid & 31;
    const int wave = tid >> 5;          // 8 waves
    const int mw   = wave >> 2;         // 0..1 -> 64-row slab
    const int nw   = wave & 3;          // 0..3 -> 32-col slab
    const int lm   = lane & 15;         // row/col within 16x16 tile
    const int g    = lane >> 4;         // lane group (K-half select)
    const int row0 = blockIdx.y * BM;
    const int col0 = blockIdx.x * BN;

    FragC acc[4][2];
    #pragma unroll
    for (int mt = 0; mt < 4; ++mt)
        #pragma unroll
        for (int nt = 0; nt < 2; ++nt)
            #pragma unroll
            for (int j = 0; j < 8; ++j) acc[mt][nt].f[j] = 0.0f;

#if TDM_AVAILABLE
    // -------- TDM double-buffered pipeline: DMA tiles to LDS, overlap with WMMA ------
    const uint32_t ldsA0 = (uint32_t)(uintptr_t)&sA[0][0][0];
    const uint32_t ldsA1 = (uint32_t)(uintptr_t)&sA[1][0][0];
    const uint32_t ldsB0 = (uint32_t)(uintptr_t)&sB[0][0][0];
    const uint32_t ldsB1 = (uint32_t)(uintptr_t)&sB[1][0][0];

    if (wave == 0) {
        tdm_load_tile_2d(ldsA0, A  + (size_t)row0 * K, BK, BM, (uint32_t)K);
        tdm_load_tile_2d(ldsB0, Bt + (size_t)col0 * K, BK, BN, (uint32_t)K);
        __builtin_amdgcn_s_wait_tensorcnt(0);
    }
    __syncthreads();

    int buf = 0;
    for (int kk = 0; kk < K; kk += BK) {
        const bool more = (kk + BK) < K;
        if (more && wave == 0) {   // prefetch next K-slab into the other buffer
            tdm_load_tile_2d(buf ? ldsA0 : ldsA1, A  + (size_t)row0 * K + kk + BK, BK, BM, (uint32_t)K);
            tdm_load_tile_2d(buf ? ldsB0 : ldsB1, Bt + (size_t)col0 * K + kk + BK, BK, BN, (uint32_t)K);
        }

        // gather fragments per documented CDNA5 VGPR layouts, then WMMA
        FragAB af[4], bfr[2];
        #pragma unroll
        for (int mt = 0; mt < 4; ++mt) {
            const uint16_t* p = &sA[buf][mw * 64 + mt * 16 + lm][g * 8];
            *(uint4*)&af[mt].u[0] = *(const uint4*)p;          // K = g*8 .. g*8+7
            *(uint4*)&af[mt].u[4] = *(const uint4*)(p + 16);   // K = 16+g*8 ..
        }
        #pragma unroll
        for (int nt = 0; nt < 2; ++nt) {
            const uint16_t* p = &sB[buf][nw * 32 + nt * 16 + lm][g * 8];
            *(uint4*)&bfr[nt].u[0] = *(const uint4*)p;         // K = g*8 .. +7
            *(uint4*)&bfr[nt].u[4] = *(const uint4*)(p + 16);  // K = 16+g*8 .. +7
        }
        #pragma unroll
        for (int mt = 0; mt < 4; ++mt)
            #pragma unroll
            for (int nt = 0; nt < 2; ++nt)
                acc[mt][nt].v = __builtin_amdgcn_wmma_f32_16x16x32_bf16(
                    false, af[mt].v, false, bfr[nt].v, (short)0, acc[mt][nt].v, false, false);

        if (more) {
            if (wave == 0) __builtin_amdgcn_s_wait_tensorcnt(0);
            __syncthreads();       // publish DMA'd tile to all waves
        }
        buf ^= 1;
    }
#else
    // -------- Fallback: cooperative loads (batched loads, then batched LDS stores) ---
    for (int kk = 0; kk < K; kk += BK) {
        uint4 ra[2], rb[2];
        #pragma unroll
        for (int t = 0; t < 2; ++t) {
            int chunk = tid + t * 256;
            int r  = chunk >> 2;
            int c8 = (chunk & 3) << 3;
            ra[t] = *(const uint4*)(A  + (size_t)(row0 + r) * K + kk + c8);
            rb[t] = *(const uint4*)(Bt + (size_t)(col0 + r) * K + kk + c8);
        }
        #pragma unroll
        for (int t = 0; t < 2; ++t) {
            int chunk = tid + t * 256;
            int r  = chunk >> 2;
            int c8 = (chunk & 3) << 3;
            *(uint4*)&sA[0][r][c8] = ra[t];
            *(uint4*)&sB[0][r][c8] = rb[t];
        }
        __syncthreads();
        FragAB af[4], bfr[2];
        #pragma unroll
        for (int mt = 0; mt < 4; ++mt) {
            const uint16_t* p = &sA[0][mw * 64 + mt * 16 + lm][g * 8];
            *(uint4*)&af[mt].u[0] = *(const uint4*)p;
            *(uint4*)&af[mt].u[4] = *(const uint4*)(p + 16);
        }
        #pragma unroll
        for (int nt = 0; nt < 2; ++nt) {
            const uint16_t* p = &sB[0][nw * 32 + nt * 16 + lm][g * 8];
            *(uint4*)&bfr[nt].u[0] = *(const uint4*)p;
            *(uint4*)&bfr[nt].u[4] = *(const uint4*)(p + 16);
        }
        #pragma unroll
        for (int mt = 0; mt < 4; ++mt)
            #pragma unroll
            for (int nt = 0; nt < 2; ++nt)
                acc[mt][nt].v = __builtin_amdgcn_wmma_f32_16x16x32_bf16(
                    false, af[mt].v, false, bfr[nt].v, (short)0, acc[mt][nt].v, false, false);
        __syncthreads();
    }
#endif

    // epilogue: C tile layout -> M = j + g*8, N = lm
    #pragma unroll
    for (int mt = 0; mt < 4; ++mt)
        #pragma unroll
        for (int nt = 0; nt < 2; ++nt) {
            int gn = col0 + nw * 32 + nt * 16 + lm;
            float bv = bias ? bias[gn] : 0.0f;
            #pragma unroll
            for (int j = 0; j < 8; ++j) {
                int gm = row0 + mw * 64 + mt * 16 + j + g * 8;
                C[(size_t)gm * N + gn] = acc[mt][nt].f[j] + bv;
            }
        }
}

// ---------------- Softmax-pooling kernel: per (b,h) global_q / global_k ----------------
__device__ __forceinline__ float block_reduce_max(float v, float* red) {
    int tid = threadIdx.x;
    red[tid] = v; __syncthreads();
    for (int s = 128; s > 0; s >>= 1) { if (tid < s) red[tid] = fmaxf(red[tid], red[tid + s]); __syncthreads(); }
    float r = red[0]; __syncthreads(); return r;
}
__device__ __forceinline__ float block_reduce_sum(float v, float* red) {
    int tid = threadIdx.x;
    red[tid] = v; __syncthreads();
    for (int s = 128; s > 0; s >>= 1) { if (tid < s) red[tid] += red[tid + s]; __syncthreads(); }
    float r = red[0]; __syncthreads(); return r;
}

__global__ __launch_bounds__(256) void stats_kernel(const float* __restrict__ qkv,
                                                    const unsigned char* __restrict__ mask,
                                                    const float* __restrict__ wql,
                                                    const float* __restrict__ wkl,
                                                    float* __restrict__ gq,
                                                    float* __restrict__ gk) {
    __shared__ float sLog[SEQ];                 // 32KB
    __shared__ float sRed[256];
    __shared__ float sAcc[4][DH];
    __shared__ __align__(16) float sW[DH];
    __shared__ float sG[DH];

    const int bh = blockIdx.x, b = bh >> 3, h = bh & 7;
    const int tid = threadIdx.x;
    const float scale = 0.125f;                 // DH^-0.5
    const float NEG = -3.0e38f;
    const size_t base = (size_t)b * SEQ * QKV3 + (size_t)h * DH;
    const unsigned char* mrow = mask + (size_t)b * SEQ;
    const int dd = tid & 63, ss = tid >> 6;

    for (int phase = 0; phase < 2; ++phase) {
        const size_t off = base + (phase ? INNER : 0);   // q block or k block
        if (tid < DH) sW[tid] = phase ? (wkl[tid] * sG[tid]) : wql[tid]; // k: fold global_q in
        __syncthreads();

        // logits + max
        float lmax = NEG;
        for (int n = tid; n < SEQ; n += 256) {
            const float* p = qkv + off + (size_t)n * QKV3;
            float d = 0.0f;
            #pragma unroll
            for (int i = 0; i < DH; i += 4) {
                float4 qv = *(const float4*)(p + i);
                float4 wv = *(const float4*)(&sW[i]);
                d += qv.x * wv.x + qv.y * wv.y + qv.z * wv.z + qv.w * wv.w;
            }
            float lg = mrow[n] ? d * scale : NEG;
            sLog[n] = lg;
            lmax = fmaxf(lmax, lg);
        }
        float gmax = block_reduce_max(lmax, sRed);

        // exp + sum
        float lsum = 0.0f;
        for (int n = tid; n < SEQ; n += 256) {
            float e = __expf(sLog[n] - gmax);
            sLog[n] = e;
            lsum += e;
        }
        float denom = block_reduce_sum(lsum, sRed);
        __syncthreads();

        // weighted pooling of raw q (or raw k): 256 threads = 4 strips x 64 dims
        float a = 0.0f;
        for (int n = ss; n < SEQ; n += 4)
            a += sLog[n] * qkv[off + (size_t)n * QKV3 + dd];
        sAcc[ss][dd] = a;
        __syncthreads();
        if (tid < DH) {
            float gv = (sAcc[0][tid] + sAcc[1][tid] + sAcc[2][tid] + sAcc[3][tid]) / denom;
            sG[tid] = gv;
            (phase ? gk : gq)[(size_t)bh * DH + tid] = gv;
        }
        __syncthreads();
    }
}

// ---------------- u = v*global_k ; r = u@w_r + b_r + q  (bf16 out) ----------------
__global__ __launch_bounds__(256) void modv_wr_kernel(const float* __restrict__ qkv,
                                                      const float* __restrict__ gk,
                                                      const float* __restrict__ w_r,
                                                      const float* __restrict__ b_r,
                                                      uint16_t* __restrict__ rout) {
    __shared__ __align__(16) uint16_t sWr[DH][72];        // [e][d], transposed w_r, bf16
    __shared__ __align__(16) uint16_t sU[HEADS][16][72];  // per-head 16-token u tiles
    __shared__ float sGk[HEADS][DH];
    __shared__ float sBr[DH];

    const int tid  = threadIdx.x;
    const int lane = tid & 31;
    const int h    = tid >> 5;                  // wave == head
    const int row0 = blockIdx.x * 16;           // global token row (b*SEQ+n)
    const int b    = row0 >> 13;                // /SEQ

    for (int i = tid; i < DH * DH; i += 256) {
        int d = i >> 6, e = i & 63;
        sWr[e][d] = f32_to_bf16(w_r[i]);        // w_r[d][e] -> sWr[e][d]
    }
    if (tid < DH) sBr[tid] = b_r[tid];
    for (int i = tid; i < HEADS * DH; i += 256) {
        int hh = i >> 6, d = i & 63;
        sGk[hh][d] = gk[(size_t)(b * HEADS + hh) * DH + d];
    }
    __syncthreads();

    // u tile: v (cols 1024..1535) modulated by global_k
    for (int i = lane; i < 16 * DH; i += 32) {
        int t = i >> 6, d = i & 63;
        float v = qkv[(size_t)(row0 + t) * QKV3 + 2 * INNER + h * DH + d];
        sU[h][t][d] = f32_to_bf16(v * sGk[h][d]);
    }
    __syncthreads();

    const int lm = lane & 15, g = lane >> 4;
    FragC acc[4];
    #pragma unroll
    for (int nt = 0; nt < 4; ++nt)
        #pragma unroll
        for (int j = 0; j < 8; ++j) acc[nt].f[j] = 0.0f;

    #pragma unroll
    for (int kk = 0; kk < DH; kk += 32) {
        FragAB af;
        const uint16_t* ap = &sU[h][lm][kk + g * 8];
        *(uint4*)&af.u[0] = *(const uint4*)ap;
        *(uint4*)&af.u[4] = *(const uint4*)(ap + 16);
        #pragma unroll
        for (int nt = 0; nt < 4; ++nt) {
            FragAB bfr;
            const uint16_t* bp = &sWr[nt * 16 + lm][kk + g * 8];
            *(uint4*)&bfr.u[0] = *(const uint4*)bp;
            *(uint4*)&bfr.u[4] = *(const uint4*)(bp + 16);
            acc[nt].v = __builtin_amdgcn_wmma_f32_16x16x32_bf16(
                false, af.v, false, bfr.v, (short)0, acc[nt].v, false, false);
        }
    }

    // epilogue: + b_r + q residual, store bf16
    #pragma unroll
    for (int nt = 0; nt < 4; ++nt) {
        int e = nt * 16 + lm;
        #pragma unroll
        for (int j = 0; j < 8; ++j) {
            int m = j + g * 8;
            float q = qkv[(size_t)(row0 + m) * QKV3 + h * DH + e];
            float v = acc[nt].f[j] + sBr[e] + q;
            rout[(size_t)(row0 + m) * INNER + h * DH + e] = f32_to_bf16(v);
        }
    }
}

// ---------------- Host launch ----------------
extern "C" void kernel_launch(void* const* d_in, const int* in_sizes, int n_in,
                              void* d_out, int out_size, void* d_ws, size_t ws_size,
                              hipStream_t stream) {
    const float*         x      = (const float*)d_in[0];
    const unsigned char* mask   = (const unsigned char*)d_in[1];
    const float*         w_qkv  = (const float*)d_in[2];
    const float*         wql    = (const float*)d_in[3];
    const float*         wkl    = (const float*)d_in[4];
    const float*         w_r    = (const float*)d_in[5];
    const float*         b_r    = (const float*)d_in[6];
    const float*         w_out  = (const float*)d_in[7];
    const float*         b_out  = (const float*)d_in[8];
    float*               out    = (float*)d_out;

    char* ws = (char*)d_ws;
    uint16_t* x_bf   = (uint16_t*)ws;  ws += (size_t)MTOK * DIM * 2;        // 64 MB
    uint16_t* wqkvT  = (uint16_t*)ws;  ws += (size_t)QKV3 * DIM * 2;        // 3 MB
    float*    qkv    = (float*)ws;     ws += (size_t)MTOK * QKV3 * 4;       // 192 MB
    uint16_t* r_bf   = (uint16_t*)ws;  ws += (size_t)MTOK * INNER * 2;      // 32 MB
    uint16_t* woutT  = (uint16_t*)ws;  ws += (size_t)DIM * INNER * 2;       // 1 MB
    float*    gq     = (float*)ws;     ws += (size_t)BATCH * HEADS * DH * 4;
    float*    gk     = (float*)ws;     ws += (size_t)BATCH * HEADS * DH * 4;

    cvt_bf16  <<<4096, 256, 0, stream>>>(x, x_bf, MTOK * DIM);
    cvt_bf16_T<<<1024, 256, 0, stream>>>(w_qkv, wqkvT, DIM, QKV3);
    cvt_bf16_T<<<512,  256, 0, stream>>>(w_out, woutT, INNER, DIM);

    // qkv = x @ w_qkv
    gemm_bf16_nt<<<dim3(QKV3 / BN, MTOK / BM), 256, 0, stream>>>(
        x_bf, wqkvT, nullptr, qkv, MTOK, QKV3, DIM);

    // global_q / global_k per (b,h)
    stats_kernel<<<BATCH * HEADS, 256, 0, stream>>>(qkv, mask, wql, wkl, gq, gk);

    // r = (v*gk)@w_r + b_r + q   (bf16)
    modv_wr_kernel<<<MTOK / 16, 256, 0, stream>>>(qkv, gk, w_r, b_r, r_bf);

    // out = r @ w_out + b_out
    gemm_bf16_nt<<<dim3(DIM / BN, MTOK / BM), 256, 0, stream>>>(
        r_bf, woutT, b_out, out, MTOK, DIM, INNER);
}
